// RibonanzaNetEncoder_17325898072626
// MI455X (gfx1250) — compile-verified, matrix operand order checked by reading, not run on hardware
//
#include <hip/hip_runtime.h>
#include <stdint.h>

#define LSEQ 256
#define P    128
#define NH   4
#define HD   32

typedef __attribute__((ext_vector_type(16))) __bf16   v16bf;
typedef __attribute__((ext_vector_type(8)))  float    v8f;
typedef __attribute__((ext_vector_type(4)))  uint32_t u32x4;
typedef __attribute__((ext_vector_type(4)))  float    f32x4;

union Frag {
  v16bf bf;
  u32x4 q[2];
};

__device__ __forceinline__ uint16_t f2bf(float f) {
  uint32_t u = __builtin_bit_cast(uint32_t, f);
  u += 0x7FFFu + ((u >> 16) & 1u);
  return (uint16_t)(u >> 16);
}

__device__ __forceinline__ v8f vzero8() {
  v8f z = {0.f, 0.f, 0.f, 0.f, 0.f, 0.f, 0.f, 0.f};
  return z;
}

// contiguous 32B (16 bf16) fragment load
__device__ __forceinline__ Frag load_frag_c(const uint16_t* p) {
  Frag f;
  f.q[0] = *(const u32x4*)(p);
  f.q[1] = *(const u32x4*)(p + 8);
  return f;
}
// two-run fragment load: 8 bf16 at p0, 8 bf16 at p1 (A/B fragment pattern)
__device__ __forceinline__ Frag load_frag2(const uint16_t* p0, const uint16_t* p1) {
  Frag f;
  f.q[0] = *(const u32x4*)(p0);
  f.q[1] = *(const u32x4*)(p1);
  return f;
}

#define WMMA_BF16(a, b, c) \
  __builtin_amdgcn_wmma_f32_16x16x32_bf16(false, (a), false, (b), (short)0, (c), false, false)

// LDS byte offset of a generic pointer that refers to __shared__ memory
typedef __attribute__((address_space(3))) uint8_t lds_u8;
__device__ __forceinline__ uint32_t lds_off_u32(const void* p) {
  return (uint32_t)(uintptr_t)(lds_u8*)p;
}

// ---------------------------------------------------------------------------
// Kernel 0: pack 5 weight matrices (128x128 f32, row-major, col = out feature)
// into bf16 WMMA B-fragment order: [mat][ks*8+nt][lane][16]
// ---------------------------------------------------------------------------
__global__ __launch_bounds__(256) void pack_weights_kernel(
    const float* __restrict__ Wq, const float* __restrict__ Wk,
    const float* __restrict__ Wv, const float* __restrict__ Wg,
    const float* __restrict__ Wo, uint16_t* __restrict__ wp) {
  const float* srcs[5] = {Wq, Wk, Wv, Wg, Wo};
  const float* W = srcs[blockIdx.x];
  uint16_t* dst = wp + (size_t)blockIdx.x * (P * P);
  for (int e = threadIdx.x; e < P * P; e += 256) {
    int f    = e >> 9;        // fragment = ks*8 + nt
    int idx  = e & 511;
    int lane = idx >> 4;
    int t    = idx & 15;
    int ks = f >> 3, nt = f & 7;
    int j = t >> 1, w = t & 1;
    int h8 = (lane >> 4) * 8;
    int K  = (j < 4) ? (h8 + 2 * j + w) : (16 + h8 + 2 * (j - 4) + w);
    int kk = ks * 32 + K;
    int nn = nt * 16 + (lane & 15);
    dst[e] = f2bf(W[kk * P + nn]);
  }
}

// ---------------------------------------------------------------------------
// Kernel 1: LayerNorm + Q/K/V/Gate projections (bf16 WMMA) + pair bias (VALU)
// 64 rows of the flattened [65536][128] per block, 256 threads (8 waves).
// ---------------------------------------------------------------------------
__global__ __launch_bounds__(256) void ln_proj_kernel(
    const float* __restrict__ x_in, const float* __restrict__ gamma,
    const float* __restrict__ beta, const float* __restrict__ Wb,
    const float* __restrict__ bg, const uint16_t* __restrict__ wp,
    uint16_t* __restrict__ qh, uint16_t* __restrict__ kh,
    uint16_t* __restrict__ vth, float* __restrict__ biasb,
    float* __restrict__ gateb) {
  __shared__ __align__(16) uint16_t xs[64][P];  // layernormed x, bf16 (16 KB)
  const int tid = threadIdx.x;
  const int blk = blockIdx.x;

  // ---- Phase A: layernorm (4 lanes per row) + bias head proj + bf16 to LDS
  {
    int r = tid >> 2, qq = tid & 3;
    long pos = (long)blk * 64 + r;
    const float* xr = x_in + pos * P + qq * 32;
    float v[32];
    float s = 0.f, s2 = 0.f;
#pragma unroll
    for (int t = 0; t < 8; ++t) {
      f32x4 d = *(const f32x4*)(xr + 4 * t);
#pragma unroll
      for (int e = 0; e < 4; ++e) {
        float x = d[e];
        v[4 * t + e] = x;
        s += x;
        s2 += x * x;
      }
    }
    s  += __shfl_xor(s, 1, 4);  s  += __shfl_xor(s, 2, 4);
    s2 += __shfl_xor(s2, 1, 4); s2 += __shfl_xor(s2, 2, 4);
    float mu  = s * (1.f / 128.f);
    float var = s2 * (1.f / 128.f) - mu * mu;
    float rs  = rsqrtf(var + 1e-5f);
    float b0 = 0.f, b1 = 0.f, b2 = 0.f, b3 = 0.f;
#pragma unroll
    for (int t = 0; t < 32; ++t) {
      int c = qq * 32 + t;
      float n = (v[t] - mu) * rs * gamma[c] + beta[c];
      v[t] = n;
      f32x4 wb = *(const f32x4*)(Wb + c * 4);  // Wb is [128][4] row-major
      b0 += n * wb[0]; b1 += n * wb[1]; b2 += n * wb[2]; b3 += n * wb[3];
    }
#pragma unroll
    for (int t = 0; t < 32; t += 2) {
      uint32_t pk = (uint32_t)f2bf(v[t]) | ((uint32_t)f2bf(v[t + 1]) << 16);
      *(uint32_t*)&xs[r][qq * 32 + t] = pk;
    }
    b0 += __shfl_xor(b0, 1, 4); b0 += __shfl_xor(b0, 2, 4);
    b1 += __shfl_xor(b1, 1, 4); b1 += __shfl_xor(b1, 2, 4);
    b2 += __shfl_xor(b2, 1, 4); b2 += __shfl_xor(b2, 2, 4);
    b3 += __shfl_xor(b3, 1, 4); b3 += __shfl_xor(b3, 2, 4);
    if (qq == 0) {
      biasb[0 * 65536 + pos] = b0;
      biasb[1 * 65536 + pos] = b1;
      biasb[2 * 65536 + pos] = b2;
      biasb[3 * 65536 + pos] = b3;
    }
  }
  __syncthreads();

  // ---- Phase B: [64x128] x [128x128] bf16 WMMA GEMM, 4 matrices
  const int wave = tid >> 5, lane = tid & 31;
  const int lane15 = lane & 15, h8 = (lane >> 4) * 8;
  for (int task = wave; task < 16; task += 8) {
    int mat = task >> 2, mt = task & 3;
    v8f c[8];
#pragma unroll
    for (int n = 0; n < 8; ++n) c[n] = vzero8();
#pragma unroll
    for (int ks = 0; ks < 4; ++ks) {
      int m = mt * 16 + lane15;
      Frag a = load_frag2(&xs[m][ks * 32 + h8], &xs[m][ks * 32 + 16 + h8]);
      const uint16_t* wb = wp + (size_t)mat * (P * P) + (ks * 8) * 512 + lane * 16;
      Frag bcur = load_frag_c(wb);
#pragma unroll
      for (int nt = 0; nt < 8; ++nt) {
        Frag bnext = bcur;
        if (nt < 7) bnext = load_frag_c(wb + (nt + 1) * 512);  // one-ahead
        c[nt] = WMMA_BF16(a.bf, bcur.bf, c[nt]);
        bcur = bnext;
      }
    }
    // epilogue: C element (m = r + 8*(lane/16), n = lane%16) per CDNA5 layout
    int mbase = mt * 16 + 8 * (lane >> 4);
#pragma unroll
    for (int nt = 0; nt < 8; ++nt) {
      int col = nt * 16 + lane15;
#pragma unroll
      for (int r = 0; r < 8; ++r) {
        long pos = (long)blk * 64 + mbase + r;
        float val = c[nt][r];
        int hh = col >> 5, dd = col & 31;
        long ii = pos >> 8, jj = pos & 255;
        if (mat == 0) {
          qh[(((long)hh * 256 + ii) * 256 + jj) * 32 + dd] = f2bf(val);
        } else if (mat == 1) {
          kh[(((long)hh * 256 + ii) * 256 + jj) * 32 + dd] = f2bf(val);
        } else if (mat == 2) {
          vth[(((long)hh * 256 + ii) * 32 + dd) * 256 + jj] = f2bf(val);
        } else {
          float g = 1.f / (1.f + __expf(-(val + bg[col])));
          gateb[pos * P + col] = g;
        }
      }
    }
  }
}

// ---------------------------------------------------------------------------
// Kernel 2: row-axis attention. One block per (h,i).
// K and V^T slabs (16KB each) are staged into LDS with CDNA5 async
// global->LDS b128 copies (ASYNCcnt), then each wave owns 2 j-tiles.
// scores = Q Kt / sqrt(32) + bias[h][j][k]; softmax over k; ctx = P V.
// ---------------------------------------------------------------------------
__global__ __launch_bounds__(256) void attn_kernel(
    const uint16_t* __restrict__ qh, const uint16_t* __restrict__ kh,
    const uint16_t* __restrict__ vth, const float* __restrict__ biasb,
    float* __restrict__ ctxb) {
  __shared__ __align__(16) uint16_t Ksl[256 * 32];      // 16 KB
  __shared__ __align__(16) uint16_t Vsl[32 * 256];      // 16 KB
  __shared__ __align__(16) uint16_t Ps[8][2][16][32];   // 16 KB (double buffer)
  const int h = blockIdx.x >> 8, i = blockIdx.x & 255;
  const int tid = threadIdx.x;
  const int wave = tid >> 5, lane = tid & 31;
  const int lane15 = lane & 15, h8 = (lane >> 4) * 8;
  const uint16_t* Qs = qh + ((long)(h * 256 + i)) * 256 * 32;
  const uint16_t* Kg = kh + ((long)(h * 256 + i)) * 256 * 32;
  const uint16_t* Vg = vth + ((long)(h * 256 + i)) * 32 * 256;
  const float* bh = biasb + (long)h * 65536;
  const float scale = 0.17677669529663687f;  // 1/sqrt(32)

  // --- async-stage K (16KB) and V^T (16KB) into LDS, tracked by ASYNCcnt ---
  {
    uint32_t lk = lds_off_u32(Ksl) + tid * 16;
    uint32_t lv = lds_off_u32(Vsl) + tid * 16;
    uint64_t gk = (uint64_t)(uintptr_t)Kg + (uint64_t)tid * 16;
    uint64_t gv = (uint64_t)(uintptr_t)Vg + (uint64_t)tid * 16;
#pragma unroll
    for (int rnd = 0; rnd < 4; ++rnd) {
      uint32_t lk2 = lk + rnd * 4096;
      uint32_t lv2 = lv + rnd * 4096;
      uint64_t gk2 = gk + rnd * 4096;
      uint64_t gv2 = gv + rnd * 4096;
      asm volatile("global_load_async_to_lds_b128 %0, %1, off"
                   :: "v"(lk2), "v"(gk2) : "memory");
      asm volatile("global_load_async_to_lds_b128 %0, %1, off"
                   :: "v"(lv2), "v"(gv2) : "memory");
    }
    asm volatile("s_wait_asynccnt 0" ::: "memory");
  }
  __syncthreads();

  for (int jt = wave; jt < 16; jt += 8) {
    // -- scores: 16 WMMAs, K-dim = head dim = 32 (single k-step)
    v8f c[16];
#pragma unroll
    for (int n = 0; n < 16; ++n) c[n] = vzero8();
    {
      int m = jt * 16 + lane15;
      Frag a = load_frag2(Qs + m * 32 + h8, Qs + m * 32 + 16 + h8);
      Frag bcur = load_frag2(&Ksl[lane15 * 32 + h8], &Ksl[lane15 * 32 + 16 + h8]);
#pragma unroll
      for (int nt = 0; nt < 16; ++nt) {
        Frag bnext = bcur;
        if (nt < 15) {
          int n = (nt + 1) * 16 + lane15;
          bnext = load_frag2(&Ksl[n * 32 + h8], &Ksl[n * 32 + 16 + h8]);
        }
        c[nt] = WMMA_BF16(a.bf, bcur.bf, c[nt]);
        bcur = bnext;
      }
    }
    // -- scale + pair bias + softmax over k (row spans 16 lanes x 16 c-regs)
    int mrow = jt * 16 + 8 * (lane >> 4);
    float inv[8];
#pragma unroll
    for (int r = 0; r < 8; ++r) {
      int jq = mrow + r;
      float rm = -3.4e38f;
#pragma unroll
      for (int nt = 0; nt < 16; ++nt) {
        int k = nt * 16 + lane15;
        float s = c[nt][r] * scale + bh[(long)jq * 256 + k];
        c[nt][r] = s;
        rm = fmaxf(rm, s);
      }
      rm = fmaxf(rm, __shfl_xor(rm, 1, 16));
      rm = fmaxf(rm, __shfl_xor(rm, 2, 16));
      rm = fmaxf(rm, __shfl_xor(rm, 4, 16));
      rm = fmaxf(rm, __shfl_xor(rm, 8, 16));
      float ssum = 0.f;
#pragma unroll
      for (int nt = 0; nt < 16; ++nt) {
        float e = __expf(c[nt][r] - rm);
        c[nt][r] = e;
        ssum += e;
      }
      ssum += __shfl_xor(ssum, 1, 16);
      ssum += __shfl_xor(ssum, 2, 16);
      ssum += __shfl_xor(ssum, 4, 16);
      ssum += __shfl_xor(ssum, 8, 16);
      inv[r] = 1.f / ssum;
    }
    // -- ctx = probs x V : C-layout -> A-layout through double-buffered
    //    per-wave LDS chunks so stores of chunk k+1 overlap WMMAs of chunk k
    v8f c2[2];
    c2[0] = vzero8();
    c2[1] = vzero8();
    int mloc = 8 * (lane >> 4);
    // chunk 0 into buffer 0
#pragma unroll
    for (int q2 = 0; q2 < 2; ++q2)
#pragma unroll
      for (int r = 0; r < 8; ++r)
        Ps[wave][0][mloc + r][q2 * 16 + lane15] = f2bf(c[q2][r] * inv[r]);
#pragma unroll
    for (int ksq = 0; ksq < 8; ++ksq) {
      if (ksq < 7) {
        int bn = (ksq + 1) & 1;
#pragma unroll
        for (int q2 = 0; q2 < 2; ++q2) {
          int nt = (ksq + 1) * 2 + q2;
#pragma unroll
          for (int r = 0; r < 8; ++r)
            Ps[wave][bn][mloc + r][q2 * 16 + lane15] = f2bf(c[nt][r] * inv[r]);
        }
      }
      int bs = ksq & 1;
      Frag a = load_frag2(&Ps[wave][bs][lane15][h8], &Ps[wave][bs][lane15][16 + h8]);
#pragma unroll
      for (int nt2 = 0; nt2 < 2; ++nt2) {
        int d = nt2 * 16 + lane15;
        Frag b = load_frag2(&Vsl[d * 256 + ksq * 32 + h8],
                            &Vsl[d * 256 + ksq * 32 + 16 + h8]);
        c2[nt2] = WMMA_BF16(a.bf, b.bf, c2[nt2]);
      }
    }
    // -- store ctx f32 in [pos][h*32+d] layout
#pragma unroll
    for (int nt2 = 0; nt2 < 2; ++nt2) {
      int d = nt2 * 16 + lane15;
      int col = h * 32 + d;
#pragma unroll
      for (int r = 0; r < 8; ++r) {
        int jq = mrow + r;
        long pos = (long)i * 256 + jq;
        ctxb[pos * P + col] = c2[nt2][r];
      }
    }
  }
}

// ---------------------------------------------------------------------------
// Kernel 3: out = (gate * ctx) @ Wo + bo + residual
// ---------------------------------------------------------------------------
__global__ __launch_bounds__(256) void out_proj_kernel(
    const float* __restrict__ ctxb, const float* __restrict__ gateb,
    const uint16_t* __restrict__ wp, const float* __restrict__ bo,
    const float* __restrict__ x_in, float* __restrict__ out) {
  __shared__ __align__(16) uint16_t as[64][P];
  const int tid = threadIdx.x, blk = blockIdx.x;
  {
    int r = tid >> 2, qq = tid & 3;
    long pos = (long)blk * 64 + r;
    const float* cp = ctxb + pos * P + qq * 32;
    const float* gp = gateb + pos * P + qq * 32;
#pragma unroll
    for (int t = 0; t < 8; ++t) {
      f32x4 cv = *(const f32x4*)(cp + 4 * t);
      f32x4 gv = *(const f32x4*)(gp + 4 * t);
      uint32_t p0 = (uint32_t)f2bf(cv[0] * gv[0]) | ((uint32_t)f2bf(cv[1] * gv[1]) << 16);
      uint32_t p1 = (uint32_t)f2bf(cv[2] * gv[2]) | ((uint32_t)f2bf(cv[3] * gv[3]) << 16);
      *(uint32_t*)&as[r][qq * 32 + 4 * t]     = p0;
      *(uint32_t*)&as[r][qq * 32 + 4 * t + 2] = p1;
    }
  }
  __syncthreads();
  const int wave = tid >> 5, lane = tid & 31;
  const int lane15 = lane & 15, h8 = (lane >> 4) * 8;
  const int mt = wave & 3, nh = wave >> 2;
  v8f c[4];
#pragma unroll
  for (int n = 0; n < 4; ++n) c[n] = vzero8();
#pragma unroll
  for (int ks = 0; ks < 4; ++ks) {
    int m = mt * 16 + lane15;
    Frag a = load_frag2(&as[m][ks * 32 + h8], &as[m][ks * 32 + 16 + h8]);
    const uint16_t* wb = wp + (size_t)4 * (P * P) + (ks * 8 + nh * 4) * 512 + lane * 16;
    Frag bcur = load_frag_c(wb);
#pragma unroll
    for (int nt = 0; nt < 4; ++nt) {
      Frag bnext = bcur;
      if (nt < 3) bnext = load_frag_c(wb + (nt + 1) * 512);
      c[nt] = WMMA_BF16(a.bf, bcur.bf, c[nt]);
      bcur = bnext;
    }
  }
  int mbase = mt * 16 + 8 * (lane >> 4);
#pragma unroll
  for (int nt = 0; nt < 4; ++nt) {
    int col = nh * 64 + nt * 16 + lane15;
    float bov = bo[col];
#pragma unroll
    for (int r = 0; r < 8; ++r) {
      long pos = (long)blk * 64 + mbase + r;
      out[pos * P + col] = c[nt][r] + bov + x_in[pos * P + col];
    }
  }
}

// ---------------------------------------------------------------------------
extern "C" void kernel_launch(void* const* d_in, const int* in_sizes, int n_in,
                              void* d_out, int out_size, void* d_ws, size_t ws_size,
                              hipStream_t stream) {
  const float* x   = (const float*)d_in[0];
  const float* gam = (const float*)d_in[1];
  const float* bet = (const float*)d_in[2];
  const float* Wq  = (const float*)d_in[3];
  const float* Wk  = (const float*)d_in[4];
  const float* Wv  = (const float*)d_in[5];
  const float* Wb  = (const float*)d_in[6];
  const float* Wg  = (const float*)d_in[7];
  const float* bg  = (const float*)d_in[8];
  const float* Wo  = (const float*)d_in[9];
  const float* bo  = (const float*)d_in[10];
  float* out = (float*)d_out;

  char* ws = (char*)d_ws;
  const size_t SZ_WPACK = 5ull * 128 * 128 * 2;       // 160 KB
  const size_t SZ_HEAD  = 4ull * 256 * 256 * 32 * 2;  // 16 MB each
  const size_t SZ_BIAS  = 4ull * 256 * 256 * 4;       // 1 MB
  const size_t SZ_FULL  = 65536ull * 128 * 4;         // 32 MB each
  uint16_t* wp    = (uint16_t*)(ws);
  uint16_t* qh    = (uint16_t*)(ws + SZ_WPACK);
  uint16_t* kh    = (uint16_t*)(ws + SZ_WPACK + SZ_HEAD);
  uint16_t* vth   = (uint16_t*)(ws + SZ_WPACK + 2 * SZ_HEAD);
  float*    biasb = (float*)   (ws + SZ_WPACK + 3 * SZ_HEAD);
  float*    gateb = (float*)   (ws + SZ_WPACK + 3 * SZ_HEAD + SZ_BIAS);
  float*    ctxb  = (float*)   (ws + SZ_WPACK + 3 * SZ_HEAD + SZ_BIAS + SZ_FULL);

  pack_weights_kernel<<<5, 256, 0, stream>>>(Wq, Wk, Wv, Wg, Wo, wp);
  ln_proj_kernel<<<1024, 256, 0, stream>>>(x, gam, bet, Wb, bg, wp, qh, kh, vth,
                                           biasb, gateb);
  attn_kernel<<<1024, 256, 0, stream>>>(qh, kh, vth, biasb, ctxb);
  out_proj_kernel<<<1024, 256, 0, stream>>>(ctxb, gateb, wp, bo, x, out);
}
